// KolmogorovArnoldPolicyNetwork_11819749999196
// MI455X (gfx1250) — matrix operand order repaired and checked by reading, not compiled, not run on hardware
//
#include <hip/hip_runtime.h>
#include <hip/hip_bf16.h>

// ---------------------------------------------------------------------------
// KAN forward for MI455X (gfx1250, wave32).
//   layer1 (B=16384, IN=1024, OUT=5): v_wmma_f32_16x16x32_bf16, fp32 accum,
//     B-fragments software-pipelined from LDS, dual accumulators.
//   layer2/3 + softmax: fp32 VALU, one thread per row (tiny work).
// ---------------------------------------------------------------------------

typedef __bf16 v16bf __attribute__((ext_vector_type(16)));
typedef float  v8f   __attribute__((ext_vector_type(8)));

#define IN1   1024
#define NB    10          // basis functions per input
#define NF    11          // features per input: silu + NB bases
#define NOUT1 5
#define NCOLS 6                         // 5 real weight columns + 1 zero column
#define COLSZ (NF * IN1)                // elements per column
#define WSUM  (NOUT1 * COLSZ)           // real weights in LDS (bf16 elements)
#define WTOT  (NCOLS * COLSZ)           // incl. zero column
#define LDSB  (WTOT * 2)                // dynamic LDS bytes (132 KB)

// Uniform quintic B-spline features. knots_m = 0.4*m - 3, m=0..15.
// The 6 nonzero degree-5 bases at cell c are universal polynomials of
// u = frac((x+3)/0.4); dense 10-slot vector = window shifted up by c.
__device__ __forceinline__ void kan_features(float xx, float* feat) {
    // silu
    float sig = 1.0f / (1.0f + __expf(-xx));
    feat[0] = xx * sig;

    float t  = fmaf(xx, 2.5f, 7.5f);        // (x + 3) / H
    float cf = floorf(t);
    float u  = t - cf;
    // cell index for the shift; 15 => everything shifts out => all-zero bases
    int cs = (cf >= 0.0f && cf < 15.0f) ? (int)cf : 15;

    float u2 = u * u;
    float u4 = u2 * u2;
    float u5 = u4 * u;
    // cardinal quintic B-spline pieces (x120)
    float p0 = ((((-u        + 5.0f) * u - 10.0f) * u + 10.0f) * u -  5.0f) * u +  1.0f; // (1-u)^5
    float p1 = (((( 5.0f * u - 20.0f) * u + 20.0f) * u + 20.0f) * u - 50.0f) * u + 26.0f;
    float p2 = ((((-10.0f * u + 30.0f) * u) * u - 60.0f) * u) * u + 66.0f;
    float p3 = (((( 10.0f * u - 20.0f) * u - 20.0f) * u + 20.0f) * u + 50.0f) * u + 26.0f;
    float p4 = ((((-5.0f * u +  5.0f) * u + 10.0f) * u + 10.0f) * u +  5.0f) * u +  1.0f;
    const float inv120 = (1.0f / 120.0f);

    // 16-slot window: E[0..5] = w, rest 0; shift up by cs; read slots 5..14.
    float E[16];
    E[0] = p0 * inv120;
    E[1] = p1 * inv120;
    E[2] = p2 * inv120;
    E[3] = p3 * inv120;
    E[4] = p4 * inv120;
    E[5] = u5 * inv120;
#pragma unroll
    for (int m = 6; m < 16; ++m) E[m] = 0.0f;

#pragma unroll
    for (int bit = 8; bit >= 1; bit >>= 1) {
        bool sel = (cs & bit) != 0;
#pragma unroll
        for (int m = 15; m >= 0; --m) {
            float src = (m >= bit) ? E[m - bit] : 0.0f;
            E[m] = sel ? src : E[m];
        }
    }
#pragma unroll
    for (int k = 0; k < NB; ++k) feat[1 + k] = E[5 + k];
}

// ---------------------------------------------------------------------------
// Kernel A: layer 1 via WMMA. 128 threads = 4 waves, 16 rows/wave, 64 rows/WG.
// h1 written into d_out columns 0..4 (row stride 64).
// ---------------------------------------------------------------------------
__global__ __launch_bounds__(128)
void kan_layer1_wmma(const float* __restrict__ x,
                     const float* __restrict__ Wb1,
                     const float* __restrict__ Ws1,
                     float* __restrict__ out /* [B][64], cols 0..4 = h1 */) {
    extern __shared__ char smem_raw[];
    __bf16* Wlds = (__bf16*)smem_raw;   // [6][NF][1024] bf16 (col 5 = zeros)

    // Stage weights in B-fragment order: col n (0..4), feature f, input i.
    // Column 5 is zero-filled (read by padding lanes n>=5; keeps EXEC full).
    for (int idx = threadIdx.x; idx < WTOT; idx += blockDim.x) {
        float v = 0.0f;
        if (idx < WSUM) {
            int i  = idx & (IN1 - 1);
            int nf = idx >> 10;
            int f  = nf % NF;
            int n  = nf / NF;
            v = (f == 0) ? Wb1[i * NOUT1 + n]
                         : Ws1[(i * NOUT1 + n) * NB + (f - 1)];
        }
        Wlds[idx] = (__bf16)v;
    }
    __syncthreads();

    const int lane = threadIdx.x & 31;
    const int wave = threadIdx.x >> 5;
    const int mrow = lane & 15;         // A-matrix row (batch row within tile)
    const int hi   = lane >> 4;         // lane half
    const int koff = hi * 16;           // A frag K offsets: {0..15} / {16..31}
    const int ksel = hi * 16;           // B frag K offsets: {0..15} / {16..31}
    const int ncol = mrow;              // B/C column this lane owns
    const int rowBase = blockIdx.x * 64 + wave * 16;
    const float* xrow = x + (size_t)(rowBase + mrow) * IN1;

    // Per-lane constant LDS base; padding lanes point at the zero column.
    const int colSel  = (ncol < NOUT1) ? ncol : NOUT1;
    const int colBase = colSel * COLSZ + ksel;

    v8f acc0 = {};
    v8f acc1 = {};

    for (int ic = 0; ic < IN1 / 32; ++ic) {
        const int i0 = ic * 32;

        // global x loads first (longest latency)
        float4 xa = *(const float4*)(xrow + i0 + koff);
        float4 xb = *(const float4*)(xrow + i0 + koff + 4);
        float4 xc = *(const float4*)(xrow + i0 + koff + 8);
        float4 xd = *(const float4*)(xrow + i0 + koff + 12);
        if (ic + 1 < IN1 / 32) __builtin_prefetch(xrow + i0 + 32, 0, 1);

        // prime the B-fragment pipeline (constant offsets -> DS imm offsets)
        const __bf16* bbase = Wlds + colBase + i0;
        v16bf bq[3];
        bq[0] = *(const v16bf*)(bbase);
        bq[1] = *(const v16bf*)(bbase + (1 << 10));

        float xs[16] = {xa.x, xa.y, xa.z, xa.w,  xb.x, xb.y, xb.z, xb.w,
                        xc.x, xc.y, xc.z, xc.w,  xd.x, xd.y, xd.z, xd.w};

        v16bf af[NF];                   // A fragments built directly in-lane
#pragma unroll
        for (int e = 0; e < 16; ++e) {
            float feat[NF];
            kan_features(xs[e], feat);
#pragma unroll
            for (int f = 0; f < NF; ++f) af[f][e] = (__bf16)feat[f];
        }

        // WMMA chain: 2-deep B-fragment pipeline + alternating accumulators
#pragma unroll
        for (int f = 0; f < NF; ++f) {
            if (f + 2 < NF)
                bq[(f + 2) % 3] = *(const v16bf*)(bbase + ((f + 2) << 10));
            if (f & 1)
                acc1 = __builtin_amdgcn_wmma_f32_16x16x32_bf16(
                           false, af[f], false, bq[f % 3], (short)0, acc1,
                           false, false);
            else
                acc0 = __builtin_amdgcn_wmma_f32_16x16x32_bf16(
                           false, af[f], false, bq[f % 3], (short)0, acc0,
                           false, false);
        }
    }

    // C layout: VGPR v, lane L -> row v + (L>=16 ? 8 : 0), col L%16
    if (ncol < NOUT1) {
#pragma unroll
        for (int v = 0; v < 8; ++v) {
            int r = rowBase + v + hi * 8;
            out[(size_t)r * 64 + ncol] = acc0[v] + acc1[v];
        }
    }
}

// ---------------------------------------------------------------------------
// Kernel B: layers 2, 3 + softmax. One thread per batch row, fp32.
// Reads h1 from d_out cols 0..4, overwrites the full 64-wide row.
// ---------------------------------------------------------------------------
__global__ __launch_bounds__(256)
void kan_tail(const float* __restrict__ Wb2, const float* __restrict__ Ws2,
              const float* __restrict__ Wb3, const float* __restrict__ Ws3,
              float* __restrict__ out, int nrows) {
    int b = blockIdx.x * blockDim.x + threadIdx.x;
    if (b >= nrows) return;

    float h1[5];
#pragma unroll
    for (int i = 0; i < 5; ++i) h1[i] = out[(size_t)b * 64 + i];

    // layer 2: 5 -> 5
    float h2[5] = {0.f, 0.f, 0.f, 0.f, 0.f};
#pragma unroll
    for (int i = 0; i < 5; ++i) {
        float feat[NF];
        kan_features(h1[i], feat);
#pragma unroll
        for (int o = 0; o < 5; ++o) {
            float a = feat[0] * Wb2[i * 5 + o];
#pragma unroll
            for (int k = 0; k < NB; ++k)
                a = fmaf(feat[1 + k], Ws2[(i * 5 + o) * NB + k], a);
            h2[o] += a;
        }
    }

    // layer 3: 5 -> 64
    float h3[64];
#pragma unroll
    for (int o = 0; o < 64; ++o) h3[o] = 0.0f;
#pragma unroll
    for (int i = 0; i < 5; ++i) {
        float feat[NF];
        kan_features(h2[i], feat);
#pragma unroll
        for (int o = 0; o < 64; ++o) {
            float a = fmaf(feat[0], Wb3[i * 64 + o], h3[o]);
#pragma unroll
            for (int k = 0; k < NB; ++k)
                a = fmaf(feat[1 + k], Ws3[(i * 64 + o) * NB + k], a);
            h3[o] = a;
        }
    }

    // softmax
    float m = h3[0];
#pragma unroll
    for (int o = 1; o < 64; ++o) m = fmaxf(m, h3[o]);
    float s = 0.0f;
#pragma unroll
    for (int o = 0; o < 64; ++o) { h3[o] = __expf(h3[o] - m); s += h3[o]; }
    float inv = 1.0f / s;
#pragma unroll
    for (int o = 0; o < 64; ++o) out[(size_t)b * 64 + o] = h3[o] * inv;
}

// ---------------------------------------------------------------------------
extern "C" void kernel_launch(void* const* d_in, const int* in_sizes, int n_in,
                              void* d_out, int out_size, void* d_ws, size_t ws_size,
                              hipStream_t stream) {
    const float* x   = (const float*)d_in[0];
    const float* Wb1 = (const float*)d_in[1];
    const float* Ws1 = (const float*)d_in[2];
    const float* Wb2 = (const float*)d_in[3];
    const float* Ws2 = (const float*)d_in[4];
    const float* Wb3 = (const float*)d_in[5];
    const float* Ws3 = (const float*)d_in[6];
    float* out = (float*)d_out;

    const int nrows = in_sizes[0] / IN1;          // 16384
    const int gridA = nrows / 64;                 // 64 rows per WG (4 waves)
    const int gridB = (nrows + 255) / 256;

    kan_layer1_wmma<<<gridA, 128, LDSB, stream>>>(x, Wb1, Ws1, out);
    kan_tail<<<gridB, 256, 0, stream>>>(Wb2, Ws2, Wb3, Ws3, out, nrows);
}
